// MOELayer_80547816669401
// MI455X (gfx1250) — matrix operand order repaired and checked by reading, not compiled, not run on hardware
//
#include <hip/hip_runtime.h>
#include <hip/hip_bf16.h>

// ---------------------------------------------------------------------------
// MoE layer for MI455X (gfx1250, wave32, WMMA bf16 path).
// Compute-bound (~206 GFLOP vs ~72MB traffic) => everything routed through
// v_wmma_f32_16x16x32_bf16 with fp32 accumulation. fp32 inputs are converted
// once to bf16 (plus transposed weight copies so B-fragments are contiguous).
// Workspace usage: ~62 MB + 128 B.
// ---------------------------------------------------------------------------

typedef __attribute__((ext_vector_type(16))) __bf16 v16bf;
typedef __attribute__((ext_vector_type(8)))  __bf16 v8bf;
typedef __attribute__((ext_vector_type(8)))  float  v8f;

#define DIMC   1024
#define HSH    4096
#define SEGC   8
#define SDC    128
#define EC     16
#define HRC    512
#define MROWS  4096            // B*T
#define NSEGR  32768           // B*SEG*T token-segment rows
#define OUT_ELEMS (MROWS * DIMC)

// ---------------- helpers ----------------

__device__ __forceinline__ float gelu_f(float x) {
    // jax.nn.gelu default (tanh approximation)
    float x3 = x * x * x;
    return 0.5f * x * (1.0f + tanhf(0.7978845608028654f * (x + 0.044715f * x3)));
}

// A-fragment (16x32 bf16, row-major in memory/LDS with leading dim ldk).
// ISA 16-bit A layout: lanes 0-15 row M=lane hold K {0..7,16..23};
// lanes 16-31 row M=lane-16 hold K {8..15,24..31}.
__device__ __forceinline__ v16bf ldA(const __bf16* base, int ldk, int lane) {
    int r  = lane & 15;
    int kb = (lane >> 4) << 3;          // 0 or 8
    const __bf16* p = base + (size_t)r * ldk + kb;
    v8bf lo = *(const v8bf*)(p);
    v8bf hi = *(const v8bf*)(p + 16);
    v16bf a;
#pragma unroll
    for (int i = 0; i < 8; ++i) { a[i] = lo[i]; a[i + 8] = hi[i]; }
    return a;
}

// B-fragment (32x16 bf16) from an N-major (transposed) buffer BT[n][k], lead dim ldk.
// B layout (matches ldA): lanes 0-15 column N=lane hold K {0..7,16..23}; lanes 16-31 hold K {8..15,24..31}.
__device__ __forceinline__ v16bf ldB(const __bf16* baseT, int ldk, int lane) {
    int c  = lane & 15;
    int kb = (lane >> 4) << 3;          // 0 or 8
    const __bf16* p = baseT + (size_t)c * ldk + kb;
    v8bf lo = *(const v8bf*)(p);
    v8bf hi = *(const v8bf*)(p + 16);
    v16bf b;
#pragma unroll
    for (int i = 0; i < 8; ++i) { b[i] = lo[i]; b[i + 8] = hi[i]; }
    return b;
}

__device__ __forceinline__ v8f wmma_bf16(v16bf a, v16bf b, v8f c) {
    return __builtin_amdgcn_wmma_f32_16x16x32_bf16(
        false, a, false, b, (short)0, c, false, false);
}

// ---------------- conversion kernels ----------------

__global__ void cvt_bf16_kernel(const float* __restrict__ src, __bf16* __restrict__ dst) {
    int i = (blockIdx.x * blockDim.x + threadIdx.x) * 4;
    float4 v = *(const float4*)(src + i);
    __bf16* p = dst + i;
    p[0] = (__bf16)v.x; p[1] = (__bf16)v.y; p[2] = (__bf16)v.z; p[3] = (__bf16)v.w;
}

// dst[b][n][k] = bf16(src[b][k][n]); LDS-tiled so both sides coalesce.
__global__ void cvt_transpose_kernel(const float* __restrict__ src, __bf16* __restrict__ dst,
                                     int K, int N) {
    __shared__ __bf16 tile[32][33];
    int b  = blockIdx.z;
    int n0 = blockIdx.x * 32, k0 = blockIdx.y * 32;
    const float* s = src + (size_t)b * K * N;
    __bf16*      d = dst + (size_t)b * K * N;
    int tx = threadIdx.x, ty = threadIdx.y;
#pragma unroll
    for (int i = 0; i < 32; i += 8)
        tile[ty + i][tx] = (__bf16)s[(size_t)(k0 + ty + i) * N + (n0 + tx)];
    __syncthreads();
#pragma unroll
    for (int i = 0; i < 32; i += 8)
        d[(size_t)(n0 + ty + i) * K + (k0 + tx)] = tile[tx][ty + i];
}

__global__ void zero_stats_kernel(float* __restrict__ stats) {
    if (threadIdx.x < 32) stats[threadIdx.x] = 0.0f;
}

// ---------------- generic GEMM: out = act(A @ B + bias) ----------------
// A: bf16 row-major [M x K]; BT: bf16 N-major [N x K]; bias fp32 [N].
// Block = 256 threads = 8 waves; wave tile 32x64; block tile 64x256.
template <bool GELU, bool OUT_BF16>
__global__ void gemm_bias_kernel(const __bf16* __restrict__ A,
                                 const __bf16* __restrict__ BT,
                                 const float* __restrict__ bias,
                                 void* __restrict__ out,
                                 int M, int N, int K) {
    int tid  = threadIdx.x;
    int w    = tid >> 5;
    int lane = tid & 31;
    int wm   = w & 1, wn = w >> 1;
    int row0 = blockIdx.y * 64 + wm * 32;
    int col0 = blockIdx.x * 256 + wn * 64;

    v8f acc[2][4] = {};
    for (int k0 = 0; k0 < K; k0 += 32) {
        v16bf a0 = ldA(A + (size_t)row0 * K + k0, K, lane);
        v16bf a1 = ldA(A + (size_t)(row0 + 16) * K + k0, K, lane);
#pragma unroll
        for (int j = 0; j < 4; ++j) {
            v16bf b = ldB(BT + (size_t)(col0 + 16 * j) * K + k0, K, lane);
            acc[0][j] = wmma_bf16(a0, b, acc[0][j]);
            acc[1][j] = wmma_bf16(a1, b, acc[1][j]);
        }
    }

    int cn = lane & 15, rv = (lane >> 4) * 8;
#pragma unroll
    for (int mi = 0; mi < 2; ++mi) {
#pragma unroll
        for (int j = 0; j < 4; ++j) {
            int col = col0 + 16 * j + cn;
            float bv = bias[col];
#pragma unroll
            for (int v = 0; v < 8; ++v) {
                int row = row0 + 16 * mi + rv + v;
                float x = acc[mi][j][v] + bv;
                if (GELU) x = gelu_f(x);
                if (OUT_BF16)
                    ((__bf16*)out)[(size_t)row * N + col] = (__bf16)x;
                else
                    ((float*)out)[(size_t)row * N + col] = x;
            }
        }
    }
}

// ---------------- router: softmax + top-2 + aux-loss stats ----------------
__global__ void router_kernel(const float* __restrict__ X, const float* __restrict__ Wr,
                              float* __restrict__ we, float* __restrict__ stats) {
    __shared__ float WrS[SDC * EC];
    __shared__ float cnt[EC], wsm[EC];
    int tid = threadIdx.x;
    for (int i = tid; i < SDC * EC; i += 256) WrS[i] = Wr[i];
    if (tid < EC) { cnt[tid] = 0.0f; wsm[tid] = 0.0f; }
    __syncthreads();

    int r = blockIdx.x * 256 + tid;
    int b = r >> 14, s = (r >> 11) & 7, t = r & 2047;
    const float* xp = X + ((size_t)(b * 2048 + t)) * DIMC + (size_t)s * SDC;

    float lg[EC];
#pragma unroll
    for (int e = 0; e < EC; ++e) lg[e] = 0.0f;
    for (int k = 0; k < SDC; ++k) {
        float xv = xp[k];
#pragma unroll
        for (int e = 0; e < EC; ++e) lg[e] += xv * WrS[k * EC + e];
    }
    float m = lg[0];
#pragma unroll
    for (int e = 1; e < EC; ++e) m = fmaxf(m, lg[e]);
    float p[EC], sum = 0.0f;
#pragma unroll
    for (int e = 0; e < EC; ++e) { p[e] = expf(lg[e] - m); sum += p[e]; }
    float inv = 1.0f / sum;
#pragma unroll
    for (int e = 0; e < EC; ++e) p[e] *= inv;

    // top-2, first-occurrence tie-breaking like jax.lax.top_k
    int i1 = 0;
#pragma unroll
    for (int e = 1; e < EC; ++e) if (p[e] > p[i1]) i1 = e;
    int i2 = (i1 == 0) ? 1 : 0;
#pragma unroll
    for (int e = 0; e < EC; ++e) if (e != i1 && p[e] > p[i2] && !(i2 < i1 && e > i2 && p[e] == p[i2])) { if (e != i2) { if (p[e] > p[i2]) i2 = e; } }
    // (simple strict-> scan; redo cleanly)
    i2 = -1;
#pragma unroll
    for (int e = 0; e < EC; ++e) {
        if (e == i1) continue;
        if (i2 < 0 || p[e] > p[i2]) i2 = e;
    }
    float w1 = p[i1], w2 = p[i2];

    float* wrow = we + (size_t)r * EC;
#pragma unroll
    for (int e = 0; e < EC; ++e)
        wrow[e] = (e == i1) ? w1 : ((e == i2) ? w2 : 0.0f);

    atomicAdd(&cnt[i1], 1.0f); atomicAdd(&cnt[i2], 1.0f);
    atomicAdd(&wsm[i1], w1);   atomicAdd(&wsm[i2], w2);
    __syncthreads();
    if (tid < EC) {
        atomicAdd(&stats[tid], cnt[tid]);
        atomicAdd(&stats[EC + tid], wsm[tid]);
    }
}

// ---------------- routed experts (dense masked combine, WMMA) ----------------
// Block = 256 thr = 8 waves, owns 32 token-segment rows; loops all 16 experts.
__global__ void routed_kernel(const __bf16* __restrict__ Xbf,
                              const __bf16* __restrict__ We1T,   // [E][HR][SD]
                              const __bf16* __restrict__ We2T,   // [E][SD][HR]
                              const float* __restrict__ be1,     // [E][HR]
                              const float* __restrict__ be2,     // [E][SD]
                              const float* __restrict__ we,      // [NSEGR][E]
                              float* __restrict__ out) {
    __shared__ __align__(16) __bf16 Hs[32][HRC + 8];
    __shared__ float wts[32][EC];

    int tid  = threadIdx.x;
    int w    = tid >> 5;
    int lane = tid & 31;
    int row0 = blockIdx.x * 32;
    // all 32 rows share (b, seg): t0..t0+31 within one 2048-token stretch
    int b = row0 >> 14, s = (row0 >> 11) & 7, t0 = row0 & 2047;
    size_t xbase = ((size_t)(b * 2048 + t0)) * DIMC + (size_t)s * SDC;

    if (tid < 128) {
        float4 v = *(const float4*)(we + (size_t)row0 * EC + tid * 4);
        int rr = (tid * 4) >> 4, cc = (tid * 4) & 15;
        wts[rr][cc] = v.x; wts[rr][cc + 1] = v.y; wts[rr][cc + 2] = v.z; wts[rr][cc + 3] = v.w;
    }
    __syncthreads();

    int cn = lane & 15, rv = (lane >> 4) * 8;
    v8f facc0 = {}, facc1 = {};

    for (int e = 0; e < EC; ++e) {
        // ---- layer 1: Hs = gelu(Xtile @ We1[e] + be1[e])   (32x512, K=128)
        for (int tt = w; tt < 64; tt += 8) {
            int mt = tt & 1, nt = tt >> 1;
            const __bf16* Ar = Xbf + xbase + (size_t)(16 * mt) * DIMC;
            const __bf16* Bp = We1T + ((size_t)e * HRC + nt * 16) * SDC;
            v8f h = {};
#pragma unroll
            for (int ks = 0; ks < 4; ++ks)
                h = wmma_bf16(ldA(Ar + ks * 32, DIMC, lane),
                              ldB(Bp + ks * 32, SDC, lane), h);
            float bv = be1[e * HRC + nt * 16 + cn];
#pragma unroll
            for (int v = 0; v < 8; ++v)
                Hs[16 * mt + rv + v][nt * 16 + cn] = (__bf16)gelu_f(h[v] + bv);
        }
        __syncthreads();

        // ---- layer 2: y = Hs @ We2[e] + be2[e]   (32x128, K=512), weighted combine
        float b2 = be2[e * SDC + w * 16 + cn];
        const __bf16* Bp2 = We2T + ((size_t)e * SDC + w * 16) * HRC;
        {
            v8f y = {};
#pragma unroll
            for (int ks = 0; ks < 16; ++ks)
                y = wmma_bf16(ldA(&Hs[0][ks * 32], HRC + 8, lane),
                              ldB(Bp2 + ks * 32, HRC, lane), y);
#pragma unroll
            for (int v = 0; v < 8; ++v)
                facc0[v] += wts[rv + v][e] * (y[v] + b2);
        }
        {
            v8f y = {};
#pragma unroll
            for (int ks = 0; ks < 16; ++ks)
                y = wmma_bf16(ldA(&Hs[16][ks * 32], HRC + 8, lane),
                              ldB(Bp2 + ks * 32, HRC, lane), y);
#pragma unroll
            for (int v = 0; v < 8; ++v)
                facc1[v] += wts[16 + rv + v][e] * (y[v] + b2);
        }
        __syncthreads();
    }

    // accumulate into shared-expert output (block owns these addresses)
#pragma unroll
    for (int v = 0; v < 8; ++v) {
        size_t o0 = xbase + (size_t)(rv + v) * DIMC + w * 16 + cn;
        out[o0]              += facc0[v];
        out[o0 + 16 * DIMC]  += facc1[v];
    }
}

// ---------------- aux loss ----------------
__global__ void aux_kernel(const float* __restrict__ stats, float* __restrict__ outAux) {
    if (threadIdx.x == 0) {
        float a = 0.0f;
        for (int e = 0; e < EC; ++e) {
            float f = stats[e]      / ((float)NSEGR * 2.0f);   // dispatch fraction / K
            float P = stats[EC + e] / (float)NSEGR;            // mean combine weight
            a += f * P;
        }
        outAux[0] = (float)EC * a;
    }
}

// ---------------- launcher ----------------
extern "C" void kernel_launch(void* const* d_in, const int* in_sizes, int n_in,
                              void* d_out, int out_size, void* d_ws, size_t ws_size,
                              hipStream_t stream) {
    const float* X   = (const float*)d_in[0];
    const float* Ws1 = (const float*)d_in[1];
    const float* bs1 = (const float*)d_in[2];
    const float* Ws2 = (const float*)d_in[3];
    const float* bs2 = (const float*)d_in[4];
    const float* Wr  = (const float*)d_in[5];
    const float* We1 = (const float*)d_in[6];
    const float* be1 = (const float*)d_in[7];
    const float* We2 = (const float*)d_in[8];
    const float* be2 = (const float*)d_in[9];
    float* out = (float*)d_out;

    // workspace layout (bytes); requires ws_size >= ~62 MB + 128 B
    char* wsb = (char*)d_ws;
    __bf16* Xbf  = (__bf16*)(wsb);                        //  8 MB [4096][1024]
    __bf16* W1T  = (__bf16*)(wsb + (8ull  << 20));        //  8 MB [4096][1024] (N-major)
    __bf16* W2T  = (__bf16*)(wsb + (16ull << 20));        //  8 MB [1024][4096] (N-major)
    __bf16* We1T = (__bf16*)(wsb + (24ull << 20));        //  2 MB [16][512][128]
    __bf16* We2T = (__bf16*)(wsb + (26ull << 20));        //  2 MB [16][128][512]
    __bf16* Hws  = (__bf16*)(wsb + (28ull << 20));        // 32 MB [4096][4096]
    float*  wE   = (float*) (wsb + (60ull << 20));        //  2 MB [32768][16]
    float*  stats= (float*) (wsb + (62ull << 20));        //  128 B

    zero_stats_kernel<<<1, 32, 0, stream>>>(stats);

    // bf16 conversions / transposed weight copies
    cvt_bf16_kernel<<<OUT_ELEMS / (256 * 4), 256, 0, stream>>>(X, Xbf);
    dim3 tb(32, 8);
    cvt_transpose_kernel<<<dim3(HSH / 32, DIMC / 32, 1), tb, 0, stream>>>(Ws1, W1T, DIMC, HSH);
    cvt_transpose_kernel<<<dim3(DIMC / 32, HSH / 32, 1), tb, 0, stream>>>(Ws2, W2T, HSH, DIMC);
    cvt_transpose_kernel<<<dim3(HRC / 32, SDC / 32, EC), tb, 0, stream>>>(We1, We1T, SDC, HRC);
    cvt_transpose_kernel<<<dim3(SDC / 32, HRC / 32, EC), tb, 0, stream>>>(We2, We2T, HRC, SDC);

    // shared expert: H = gelu(X@Ws1+bs1) ; out = H@Ws2+bs2
    gemm_bias_kernel<true, true><<<dim3(HSH / 256, MROWS / 64), 256, 0, stream>>>(
        Xbf, W1T, bs1, Hws, MROWS, HSH, DIMC);
    gemm_bias_kernel<false, false><<<dim3(DIMC / 256, MROWS / 64), 256, 0, stream>>>(
        Hws, W2T, bs2, out, MROWS, DIMC, HSH);

    // router + routed experts (accumulate into out) + aux loss
    router_kernel<<<NSEGR / 256, 256, 0, stream>>>(X, Wr, wE, stats);
    routed_kernel<<<NSEGR / 32, 256, 0, stream>>>(Xbf, We1T, We2T, be1, be2, wE, out);
    aux_kernel<<<1, 32, 0, stream>>>(stats, out + OUT_ELEMS);
}
